// Policy_42331197669623
// MI455X (gfx1250) — matrix operand order repaired and checked by reading, not compiled, not run on hardware
//
#include <hip/hip_runtime.h>
#include <hip/hip_bf16.h>

typedef __attribute__((ext_vector_type(16))) _Float16 v16h;
typedef __attribute__((ext_vector_type(8)))  float    v8f;

// d_out layout (floats): out_x[128,9] out_y[128,9] out_enum[128,3] out_delay[128,5]
//                        out_target[128,4096] h1[128,128] c1[128,128]
#define OFF_X     0
#define OFF_Y     1152
#define OFF_ENUM  2304
#define OFF_DELAY 2688
#define OFF_TGT   3328
#define OFF_H1    527616
#define OFF_C1    544000

// d_ws layout (floats)
#define WS_MAX    0        // [2][128][128] pooled maxes (side, b, n)
#define WS_ATTN   32768    // [128][128] attention vectors
#define WS_BIASC  49152    // [2][128] per-batch score bias constants
#define WS_WT     49408    // [2][128][32] folded attention weights w~ = W2^T @ attn

__device__ __forceinline__ float sigmoidf_(float x){ return 1.0f/(1.0f+__expf(-x)); }

// Build the WMMA A-fragment (16 units x K=32 f16) for one M-tile directly in
// the ISA lane layout: lane hi=0 holds K {0..7,16..23}, hi=1 holds {8..15,24..31}.
__device__ __forceinline__ v16h make_A(const float* __restrict__ bin, int unit0,
                                       int nlo, int hi,
                                       const float* __restrict__ sW1,
                                       const float* __restrict__ sb1)
{
  const int u = unit0 + nlo;
  const float i0 = bin[u*3+0], i1 = bin[u*3+1], i2 = bin[u*3+2];
  v16h A;
  #pragma unroll
  for (int e = 0; e < 16; ++e) {
    const int k = (e < 8) ? (hi*8 + e) : (16 + hi*8 + (e - 8));
    float h = fmaf(i0, sW1[k*3+0], fmaf(i1, sW1[k*3+1], fmaf(i2, sW1[k*3+2], sb1[k])));
    A[e] = (_Float16)fmaxf(h, 0.0f);
  }
  return A;
}

// Tree max over two D fragments (depth ~4, ILP-friendly: no serial chain).
__device__ __forceinline__ float max_tree16(const v8f a, const v8f b)
{
  float m0 = fmaxf(a[0], b[0]);
  float m1 = fmaxf(a[1], b[1]);
  float m2 = fmaxf(a[2], b[2]);
  float m3 = fmaxf(a[3], b[3]);
  float m4 = fmaxf(a[4], b[4]);
  float m5 = fmaxf(a[5], b[5]);
  float m6 = fmaxf(a[6], b[6]);
  float m7 = fmaxf(a[7], b[7]);
  m0 = fmaxf(m0, m4); m1 = fmaxf(m1, m5);
  m2 = fmaxf(m2, m6); m3 = fmaxf(m3, m7);
  m0 = fmaxf(m0, m2); m1 = fmaxf(m1, m3);
  return fmaxf(m0, m1);
}

// ---------------------------------------------------------------------------
// Pass 1: per-unit MLP via WMMA, max-pool over units. One workgroup per
// (batch, side); 8 waves, each wave owns 256 units and keeps all 8 B-tiles
// of W2 (f16) resident in registers. Four WMMAs in flight per step (2 M-tiles
// x 2 N-tiles) so the WMMA->VALU RAW hazard is covered by independent WMMAs
// and the other pair's tree-reduce.
// ---------------------------------------------------------------------------
__global__ __launch_bounds__(256) void unit_max_kernel(
    const float* __restrict__ enh_in, const float* __restrict__ anh_in,
    const float* __restrict__ W_enh1, const float* __restrict__ b_enh1,
    const float* __restrict__ W_enh2, const float* __restrict__ b_enh2,
    const float* __restrict__ W_anh1, const float* __restrict__ b_anh1,
    const float* __restrict__ W_anh2, const float* __restrict__ b_anh2,
    float* __restrict__ max_out)          // [2][128][128]
{
  __shared__ float sW1[96];
  __shared__ float sb1[32];
  __shared__ float wred[8][128];

  const int t    = threadIdx.x;
  const int wave = t >> 5;
  const int lane = t & 31;
  const int nlo  = lane & 15;
  const int hi   = lane >> 4;

  const int b    = blockIdx.x >> 1;
  const int side = blockIdx.x & 1;

  const float* in = side ? anh_in : enh_in;
  const float* W1 = side ? W_anh1 : W_enh1;
  const float* b1 = side ? b_anh1 : b_enh1;
  const float* W2 = side ? W_anh2 : W_enh2;
  const float* b2 = side ? b_anh2 : b_enh2;

  if (t < 96)  sW1[t] = W1[t];
  if (t < 32)  sb1[t] = b1[t];
  __syncthreads();

  // B fragments (W2^T tiles): lane nlo = column N, element e = K = e<8 ? hi*8+e : 16+hi*8+(e-8).
  v16h Bfrag[8];
  #pragma unroll
  for (int nt = 0; nt < 8; ++nt) {
    const float* w2row = W2 + (size_t)(nt*16 + nlo)*32;
    #pragma unroll
    for (int e = 0; e < 16; ++e) Bfrag[nt][e] = (_Float16)w2row[(e < 8) ? (hi*8 + e) : (16 + hi*8 + (e - 8))];
  }

  float vmax[8];
  #pragma unroll
  for (int nt = 0; nt < 8; ++nt) vmax[nt] = -3.402823466e38f;

  const float* bin = in + (size_t)b * 2048 * 3;

  for (int mt = 0; mt < 16; mt += 2) {
    const int unit0 = wave*256 + mt*16;
    v16h A0 = make_A(bin, unit0,      nlo, hi, sW1, sb1);
    v16h A1 = make_A(bin, unit0 + 16, nlo, hi, sW1, sb1);
    #pragma unroll
    for (int nt = 0; nt < 8; nt += 2) {
      v8f z0 = {}, z1 = {}, z2 = {}, z3 = {};
      v8f d00 = __builtin_amdgcn_wmma_f32_16x16x32_f16(
          false, A0, false, Bfrag[nt],   (short)0, z0, false, false);
      v8f d01 = __builtin_amdgcn_wmma_f32_16x16x32_f16(
          false, A1, false, Bfrag[nt],   (short)0, z1, false, false);
      v8f d10 = __builtin_amdgcn_wmma_f32_16x16x32_f16(
          false, A0, false, Bfrag[nt+1], (short)0, z2, false, false);
      v8f d11 = __builtin_amdgcn_wmma_f32_16x16x32_f16(
          false, A1, false, Bfrag[nt+1], (short)0, z3, false, false);
      vmax[nt]   = fmaxf(vmax[nt],   max_tree16(d00, d01));
      vmax[nt+1] = fmaxf(vmax[nt+1], max_tree16(d10, d11));
    }
  }

  // combine lane halves (rows M 0..7 vs 8..15 of each tile share n = nlo)
  #pragma unroll
  for (int nt = 0; nt < 8; ++nt)
    vmax[nt] = fmaxf(vmax[nt], __shfl_xor(vmax[nt], 16, 32));
  if (hi == 0) {
    #pragma unroll
    for (int nt = 0; nt < 8; ++nt) wred[wave][nt*16 + nlo] = vmax[nt];
  }
  __syncthreads();
  if (t < 128) {
    float m = wred[0][t];
    #pragma unroll
    for (int w = 1; w < 8; ++w) m = fmaxf(m, wred[w][t]);
    // layer-2 bias is unit-invariant: fold it in after the max
    max_out[((size_t)side*128 + b)*128 + t] = m + b2[t];
  }
}

// ---------------------------------------------------------------------------
// Fused middle stage: loc/env embed, pre+LN, LSTM cell, heads, attention,
// plus the folded score weights w~[k] = sum_n W2[n][k]*attn[n] and the
// per-batch score bias constants sum_n b2[n]*attn[n].
// One 128-thread workgroup per batch element.
// ---------------------------------------------------------------------------
__global__ __launch_bounds__(128) void mid_kernel(
    const float* __restrict__ loc, const float* __restrict__ env,
    const float* __restrict__ h0, const float* __restrict__ c0,
    const float* __restrict__ W_loc, const float* __restrict__ b_loc,
    const float* __restrict__ W_env, const float* __restrict__ b_env,
    const float* __restrict__ W_pre, const float* __restrict__ b_pre,
    const float* __restrict__ W_ih, const float* __restrict__ b_ih,
    const float* __restrict__ W_hh, const float* __restrict__ b_hh,
    const float* __restrict__ ln_g, const float* __restrict__ ln_b,
    const float* __restrict__ W_mx, const float* __restrict__ b_mx,
    const float* __restrict__ W_my, const float* __restrict__ b_my,
    const float* __restrict__ W_enum_, const float* __restrict__ b_enum_,
    const float* __restrict__ W_delay, const float* __restrict__ b_delay,
    const float* __restrict__ W_attn, const float* __restrict__ b_attn,
    const float* __restrict__ b_enh2, const float* __restrict__ b_anh2,
    const float* __restrict__ W_enh2, const float* __restrict__ W_anh2,
    const float* __restrict__ maxes,
    float* __restrict__ attn_out, float* __restrict__ biasc_out,
    float* __restrict__ wt_out, float* __restrict__ out)
{
  __shared__ float xin[512];
  __shared__ float sh0[128], xln[128], sh1[128], sattn[128];
  __shared__ float lg[26];
  __shared__ float r0[4], r1[4];

  const int b = blockIdx.x, t = threadIdx.x;
  const int wave = t >> 5, lane = t & 31;

  const float l0 = loc[b*2], l1 = loc[b*2+1];
  xin[t] = fmaxf(fmaf(l0, W_loc[t*2], fmaf(l1, W_loc[t*2+1], b_loc[t])), 0.f);
  const float e0 = env[b*2], e1 = env[b*2+1];
  xin[128+t] = fmaxf(fmaf(e0, W_env[t*2], fmaf(e1, W_env[t*2+1], b_env[t])), 0.f);
  xin[256+t] = maxes[(size_t)b*128 + t];
  xin[384+t] = maxes[(size_t)(128 + b)*128 + t];
  sh0[t] = h0[(size_t)b*128 + t];
  __syncthreads();

  // pre layer [512 -> 128] + relu
  float acc = b_pre[t];
  const float* wr = W_pre + (size_t)t*512;
  for (int k = 0; k < 512; ++k) acc = fmaf(xin[k], wr[k], acc);
  const float x = fmaxf(acc, 0.f);

  // layernorm over 128
  float s = x, s2 = x*x;
  #pragma unroll
  for (int m = 1; m < 32; m <<= 1) { s += __shfl_xor(s, m, 32); s2 += __shfl_xor(s2, m, 32); }
  if (lane == 0) { r0[wave] = s; r1[wave] = s2; }
  __syncthreads();
  const float sum  = r0[0]+r0[1]+r0[2]+r0[3];
  const float sum2 = r1[0]+r1[1]+r1[2]+r1[3];
  const float mu  = sum * (1.f/128.f);
  const float var = sum2 * (1.f/128.f) - mu*mu;
  const float xn = (x - mu) * rsqrtf(var + 1e-5f) * ln_g[t] + ln_b[t];
  xln[t] = xn;
  __syncthreads();

  // LSTM gates (i,f,g,o): thread t owns gate rows gi*128+t
  float g[4];
  #pragma unroll
  for (int gi = 0; gi < 4; ++gi) {
    const int row = gi*128 + t;
    float a = b_ih[row] + b_hh[row];
    const float* wi = W_ih + (size_t)row*128;
    const float* wh = W_hh + (size_t)row*128;
    for (int k = 0; k < 128; ++k) a = fmaf(xln[k], wi[k], fmaf(sh0[k], wh[k], a));
    g[gi] = a;
  }
  const float ig = sigmoidf_(g[0]);
  const float fg = sigmoidf_(g[1]);
  const float gg = tanhf(g[2]);
  const float og = sigmoidf_(g[3]);
  const float c1 = fg * c0[(size_t)b*128 + t] + ig * gg;
  const float h1 = og * tanhf(c1);
  out[OFF_H1 + (size_t)b*128 + t] = h1;
  out[OFF_C1 + (size_t)b*128 + t] = c1;
  sh1[t] = h1;
  __syncthreads();

  // head logits (26 rows total)
  if (t < 26) {
    const float* Wr; float bb;
    if (t < 9)       { Wr = W_mx    + t*128;      bb = b_mx[t]; }
    else if (t < 18) { Wr = W_my    + (t-9)*128;  bb = b_my[t-9]; }
    else if (t < 21) { Wr = W_enum_ + (t-18)*128; bb = b_enum_[t-18]; }
    else             { Wr = W_delay + (t-21)*128; bb = b_delay[t-21]; }
    float a = bb;
    for (int k = 0; k < 128; ++k) a = fmaf(sh1[k], Wr[k], a);
    lg[t] = a;
  }
  // attention vector
  float at = b_attn[t];
  const float* wa = W_attn + (size_t)t*128;
  for (int k = 0; k < 128; ++k) at = fmaf(sh1[k], wa[k], at);
  attn_out[(size_t)b*128 + t] = at;
  sattn[t] = at;
  __syncthreads();

  // small-head softmaxes
  if (t < 4) {
    const int lst[4] = {0,9,18,21};
    const int len[4] = {9,9,3,5};
    const int ob[4]  = {OFF_X, OFF_Y, OFF_ENUM, OFF_DELAY};
    const int s0i = lst[t], L = len[t];
    float m = lg[s0i];
    for (int i = 1; i < L; ++i) m = fmaxf(m, lg[s0i+i]);
    float se = 0.f;
    for (int i = 0; i < L; ++i) se += __expf(lg[s0i+i]-m);
    const float inv = 1.f/se;
    for (int i = 0; i < L; ++i) out[ob[t] + b*L + i] = __expf(lg[s0i+i]-m)*inv;
  }

  // folded score weights: w~[side][k] = sum_n W2[n][k] * attn[n]  (f32 exact)
  if (t < 64) {
    const int side = t >> 5, k = t & 31;
    const float* W2 = side ? W_anh2 : W_enh2;
    float a = 0.f;
    for (int n = 0; n < 128; ++n) a = fmaf(sattn[n], W2[n*32 + k], a);
    wt_out[((size_t)side*128 + b)*32 + k] = a;
  }

  // per-batch score bias constants: sum_n b2[n]*attn[n] for each side
  float pE = sattn[t]*b_enh2[t], pA = sattn[t]*b_anh2[t];
  #pragma unroll
  for (int m = 1; m < 32; m <<= 1) { pE += __shfl_xor(pE, m, 32); pA += __shfl_xor(pA, m, 32); }
  if (lane == 0) { r0[wave] = pE; r1[wave] = pA; }
  __syncthreads();
  if (t == 0) {
    biasc_out[b]       = r0[0]+r0[1]+r0[2]+r0[3];
    biasc_out[128 + b] = r1[0]+r1[1]+r1[2]+r1[3];
  }
}

// ---------------------------------------------------------------------------
// Pass 2: score[u] = hidden[u] . w~ + biasc   (W2 folded into attn, so only
// 32 MACs/unit for layer 2). One unit per lane, fully parallel, f32 exact.
// Grid: (b, side, chunk-of-256-units) = 2048 blocks x 256 threads.
// ---------------------------------------------------------------------------
__global__ __launch_bounds__(256) void score_kernel(
    const float* __restrict__ enh_in, const float* __restrict__ anh_in,
    const float* __restrict__ W_enh1, const float* __restrict__ b_enh1,
    const float* __restrict__ W_anh1, const float* __restrict__ b_anh1,
    const float* __restrict__ wt_buf,   // [2][128][32]
    const float* __restrict__ bias_c,   // [2][128]
    float* __restrict__ score_out)      // d_out target region
{
  __shared__ float sW1[96], sb1[32], swt[32];
  const int t = threadIdx.x;
  const int blk   = blockIdx.x;
  const int chunk = blk & 7;
  const int side  = (blk >> 3) & 1;
  const int b     = blk >> 4;

  const float* in = side ? anh_in : enh_in;
  const float* W1 = side ? W_anh1 : W_enh1;
  const float* b1 = side ? b_anh1 : b_enh1;

  if (t < 96) sW1[t] = W1[t];
  if (t >= 96 && t < 128) sb1[t-96] = b1[t-96];
  if (t >= 128 && t < 160) swt[t-128] = wt_buf[((size_t)side*128 + b)*32 + (t-128)];
  __syncthreads();

  const int u = chunk*256 + t;
  const float* p = in + ((size_t)b*2048 + u)*3;
  const float i0 = p[0], i1 = p[1], i2 = p[2];
  float acc = bias_c[side*128 + b];
  #pragma unroll
  for (int k = 0; k < 32; ++k) {
    float h = fmaf(i0, sW1[k*3+0], fmaf(i1, sW1[k*3+1], fmaf(i2, sW1[k*3+2], sb1[k])));
    acc = fmaf(fmaxf(h, 0.f), swt[k], acc);
  }
  score_out[(size_t)b*4096 + side*2048 + u] = acc;
}

// ---------------------------------------------------------------------------
// Softmax over 4096 target scores per batch element (in place in d_out).
// ---------------------------------------------------------------------------
__global__ __launch_bounds__(256) void target_softmax_kernel(float* __restrict__ out)
{
  __shared__ float red[8];
  const int b = blockIdx.x, t = threadIdx.x;
  const int wave = t >> 5, lane = t & 31;
  float* sc = out + OFF_TGT + (size_t)b*4096;

  float m = -3.402823466e38f;
  for (int i = t; i < 4096; i += 256) m = fmaxf(m, sc[i]);
  #pragma unroll
  for (int k = 1; k < 32; k <<= 1) m = fmaxf(m, __shfl_xor(m, k, 32));
  if (lane == 0) red[wave] = m;
  __syncthreads();
  float bm = red[0];
  #pragma unroll
  for (int w = 1; w < 8; ++w) bm = fmaxf(bm, red[w]);

  float s = 0.f;
  for (int i = t; i < 4096; i += 256) s += __expf(sc[i]-bm);
  #pragma unroll
  for (int k = 1; k < 32; k <<= 1) s += __shfl_xor(s, k, 32);
  __syncthreads();
  if (lane == 0) red[wave] = s;
  __syncthreads();
  float tot = 0.f;
  #pragma unroll
  for (int w = 0; w < 8; ++w) tot += red[w];
  const float inv = 1.f/tot;
  for (int i = t; i < 4096; i += 256) sc[i] = __expf(sc[i]-bm)*inv;
}

// ---------------------------------------------------------------------------
extern "C" void kernel_launch(void* const* d_in, const int* in_sizes, int n_in,
                              void* d_out, int out_size, void* d_ws, size_t ws_size,
                              hipStream_t stream)
{
  const float* loc    = (const float*)d_in[0];
  const float* env    = (const float*)d_in[1];
  const float* enh    = (const float*)d_in[2];
  const float* anh    = (const float*)d_in[3];
  const float* h0     = (const float*)d_in[4];
  const float* c0     = (const float*)d_in[5];
  const float* W_loc  = (const float*)d_in[6];  const float* b_loc  = (const float*)d_in[7];
  const float* W_env  = (const float*)d_in[8];  const float* b_env  = (const float*)d_in[9];
  const float* W_enh1 = (const float*)d_in[10]; const float* b_enh1 = (const float*)d_in[11];
  const float* W_enh2 = (const float*)d_in[12]; const float* b_enh2 = (const float*)d_in[13];
  const float* W_anh1 = (const float*)d_in[14]; const float* b_anh1 = (const float*)d_in[15];
  const float* W_anh2 = (const float*)d_in[16]; const float* b_anh2 = (const float*)d_in[17];
  const float* W_pre  = (const float*)d_in[18]; const float* b_pre  = (const float*)d_in[19];
  const float* W_ih   = (const float*)d_in[20]; const float* b_ih   = (const float*)d_in[21];
  const float* W_hh   = (const float*)d_in[22]; const float* b_hh   = (const float*)d_in[23];
  const float* ln_g   = (const float*)d_in[24]; const float* ln_b   = (const float*)d_in[25];
  const float* W_mx   = (const float*)d_in[26]; const float* b_mx   = (const float*)d_in[27];
  const float* W_my   = (const float*)d_in[28]; const float* b_my   = (const float*)d_in[29];
  const float* W_en   = (const float*)d_in[30]; const float* b_en   = (const float*)d_in[31];
  const float* W_dl   = (const float*)d_in[32]; const float* b_dl   = (const float*)d_in[33];
  const float* W_at   = (const float*)d_in[34]; const float* b_at   = (const float*)d_in[35];

  float* out   = (float*)d_out;
  float* ws    = (float*)d_ws;
  float* maxes = ws + WS_MAX;
  float* attnb = ws + WS_ATTN;
  float* biasc = ws + WS_BIASC;
  float* wtb   = ws + WS_WT;

  // Pass 1: unit MLP via WMMA, max-pool only (emb never materialized).
  unit_max_kernel<<<256, 256, 0, stream>>>(enh, anh, W_enh1, b_enh1, W_enh2, b_enh2,
      W_anh1, b_anh1, W_anh2, b_anh2, maxes);
  // Middle: embeds + pre + LN + LSTM + heads + attention + folded weights.
  mid_kernel<<<128, 128, 0, stream>>>(loc, env, h0, c0, W_loc, b_loc, W_env, b_env,
      W_pre, b_pre, W_ih, b_ih, W_hh, b_hh, ln_g, ln_b, W_mx, b_mx, W_my, b_my,
      W_en, b_en, W_dl, b_dl, W_at, b_at, b_enh2, b_anh2, W_enh2, W_anh2,
      maxes, attnb, biasc, wtb, out);
  // Pass 2: scores via folded weights (32 MACs/unit, pure f32 VALU).
  score_kernel<<<2048, 256, 0, stream>>>(enh, anh, W_enh1, b_enh1, W_anh1, b_anh1,
      wtb, biasc, out + OFF_TGT);
  // Softmax over 4096 scores per batch element.
  target_softmax_kernel<<<128, 256, 0, stream>>>(out);

  (void)in_sizes; (void)n_in; (void)out_size; (void)ws_size;
}
